// SelfAttentionHead_41901700939831
// MI455X (gfx1250) — matrix operand order repaired and checked by reading, not compiled, not run on hardware
//
#include <hip/hip_runtime.h>
#include <math.h>

// ---------------------------------------------------------------------------
// SelfAttentionHead (softmax over QUERY axis) for MI455X / gfx1250, wave32.
//   B=8, T=2048, E=1024, H=128
// Pipeline:
//   1) qkv_proj : q,k,v = f16(x @ W)                    (WMMA f32_16x16x32_f16)
//   2) scores   : att = sqrt(H)*q@k^T, causal mask      (WMMA, f32 out to ws)
//   3) colstats : per-column (over t) online max / 1/sum-exp, 4-way ILP
//   4) attn_out : out = softmaxed(att) @ v              (WMMA, p->f16 on the fly)
// WMMA kernels: 256 thr = 8 waves, wave owns 16x128 of C (8 accum tiles),
// ping-pong LDS double buffering, staging fully unrolled + vectorized (b128
// global traffic, no per-element loadcnt waits).
// ---------------------------------------------------------------------------

typedef __attribute__((ext_vector_type(16))) _Float16 v16h;
typedef __attribute__((ext_vector_type(8)))  _Float16 v8h;
typedef __attribute__((ext_vector_type(4)))  _Float16 v4h;
typedef __attribute__((ext_vector_type(8)))  float    v8f;

constexpr int B = 8, T = 2048, E = 1024, H = 128;
constexpr float MASK_NEG = -1.0e30f;      // finite "-inf": exp underflows to 0

static __device__ __forceinline__ v8f wmma_f16(v16h a, v16h b, v8f c) {
  // D = A(16x32,f16) * B(32x16,f16) + C(16x16,f32)
  return __builtin_amdgcn_wmma_f32_16x16x32_f16(false, a, false, b,
                                                (short)0, c, false, false);
}

// A fragment (16x32 f16): lane holds row m=lane&15, half=lane>>4.
// K indices: e<8 -> 8*half+e ; e>=8 -> 16+8*half+(e-8)  => two contiguous 16B chunks.
static __device__ __forceinline__ v16h load_a_frag(const _Float16* row, int half) {
  v8h lo = *(const v8h*)(row + 8 * half);
  v8h hi = *(const v8h*)(row + 16 + 8 * half);
  return __builtin_shufflevector(lo, hi, 0,1,2,3,4,5,6,7,8,9,10,11,12,13,14,15);
}

// B fragment (32x16 f16): lane holds col n=lane&15; K same as A: e<8 -> 8*half+e ; e>=8 -> 16+8*half+(e-8).
// LDS staged N-major with row stride 32 halves.
static __device__ __forceinline__ v16h load_b_frag(const _Float16* rown, int half) {
  v8h lo = *(const v8h*)(rown + 8 * half);
  v8h hi = *(const v8h*)(rown + 16 + 8 * half);
  return __builtin_shufflevector(lo, hi, 0,1,2,3,4,5,6,7,8,9,10,11,12,13,14,15);
}

static __device__ __forceinline__ v8f zero8() {
  v8f z = {0.f, 0.f, 0.f, 0.f, 0.f, 0.f, 0.f, 0.f};
  return z;
}

// ---------------------------------------------------------------------------
// 1) q/k/v projection: (B*T x E) @ (E x H) -> f16 (B*T x H).  grid=(B*T/128, 3)
// ---------------------------------------------------------------------------
__global__ __launch_bounds__(256) void qkv_proj(
    const float* __restrict__ x,
    const float* __restrict__ Wq, const float* __restrict__ Wk,
    const float* __restrict__ Wv,
    _Float16* __restrict__ q, _Float16* __restrict__ k,
    _Float16* __restrict__ v) {
  __shared__ _Float16 ldsX[2][128 * 32];    // [t_local][kk]   (A, row-major)
  __shared__ _Float16 ldsWt[2][128 * 32];   // [h][kk]         (B, N-major)

  const float* W;
  _Float16* out;
  if (blockIdx.y == 0)      { W = Wq; out = q; }
  else if (blockIdx.y == 1) { W = Wk; out = k; }
  else                      { W = Wv; out = v; }

  const int row0 = blockIdx.x * 128;
  const int tid = threadIdx.x, wave = tid >> 5, lane = tid & 31;
  const int m = lane & 15, half = lane >> 4;

  v8f acc[8];
#pragma unroll
  for (int i = 0; i < 8; ++i) acc[i] = zero8();

  auto stage = [&](int buf, int kk) {
    // X tile: 128x32 f32 -> f16, as 1024 float4 chunks (4 per thread).
#pragma unroll
    for (int i = 0; i < 4; ++i) {
      int chunk = tid + i * 256;                     // 0..1023
      int r = chunk >> 3, c = (chunk & 7) * 4;
      float4 xv = *(const float4*)(x + (size_t)(row0 + r) * E + kk + c);
      v4h h = {(_Float16)xv.x, (_Float16)xv.y, (_Float16)xv.z, (_Float16)xv.w};
      *(v4h*)(&ldsX[buf][r * 32 + c]) = h;
    }
    // W tile transposed: float4 loads coalesced over H, b16 scatter into LDS.
#pragma unroll
    for (int i = 0; i < 4; ++i) {
      int chunk = tid + i * 256;                     // 0..1023 over 32x128
      int r = chunk >> 5, c = (chunk & 31) * 4;
      float4 wv = *(const float4*)(W + (size_t)(kk + r) * H + c);
      ldsWt[buf][(c + 0) * 32 + r] = (_Float16)wv.x;
      ldsWt[buf][(c + 1) * 32 + r] = (_Float16)wv.y;
      ldsWt[buf][(c + 2) * 32 + r] = (_Float16)wv.z;
      ldsWt[buf][(c + 3) * 32 + r] = (_Float16)wv.w;
    }
  };

  stage(0, 0);
  __syncthreads();
  const int nSteps = E / 32;
  for (int step = 0; step < nSteps; ++step) {
    const int cur = step & 1;
    v16h a = load_a_frag(&ldsX[cur][(wave * 16 + m) * 32], half);
    v16h bf[8];
#pragma unroll
    for (int nt = 0; nt < 8; ++nt)
      bf[nt] = load_b_frag(&ldsWt[cur][(nt * 16 + m) * 32], half);
#pragma unroll
    for (int nt = 0; nt < 8; ++nt) acc[nt] = wmma_f16(a, bf[nt], acc[nt]);
    if (step + 1 < nSteps) stage(cur ^ 1, (step + 1) * 32);
    __syncthreads();
  }

#pragma unroll
  for (int nt = 0; nt < 8; ++nt)
#pragma unroll
    for (int r = 0; r < 8; ++r) {
      int M = r + 8 * half;                          // C layout: VGPR r, half
      out[(size_t)(row0 + wave * 16 + M) * H + nt * 16 + m] = (_Float16)acc[nt][r];
    }
}

// ---------------------------------------------------------------------------
// 2) scores: att[b,t,s] = sqrt(H) * q.k, masked (s>t -> MASK_NEG), f32 out.
//    grid=(T/128, T/128, B); upper-triangle tile blocks exit immediately.
// ---------------------------------------------------------------------------
__global__ __launch_bounds__(256) void scores(
    const _Float16* __restrict__ q, const _Float16* __restrict__ k,
    float* __restrict__ att) {
  const int tTile = blockIdx.x, sTile = blockIdx.y, b = blockIdx.z;
  if (sTile > tTile) return;

  __shared__ _Float16 ldsQ[2][128 * 32];
  __shared__ _Float16 ldsK[2][128 * 32];  // k rows H-contiguous => already N-major

  const int t0 = tTile * 128, s0 = sTile * 128;
  const _Float16* qb = q + (size_t)b * T * H;
  const _Float16* kb = k + (size_t)b * T * H;
  const int tid = threadIdx.x, wave = tid >> 5, lane = tid & 31;
  const int m = lane & 15, half = lane >> 4;

  v8f acc[8];
#pragma unroll
  for (int i = 0; i < 8; ++i) acc[i] = zero8();

  auto stage = [&](int buf, int kk) {
#pragma unroll
    for (int i = 0; i < 2; ++i) {                    // 512 * v8h = 128x32 halves
      int chunk = tid + i * 256;
      int r = chunk >> 2, c = (chunk & 3) * 8;
      ((v8h*)ldsQ[buf])[chunk] = *(const v8h*)(qb + (size_t)(t0 + r) * H + kk + c);
      ((v8h*)ldsK[buf])[chunk] = *(const v8h*)(kb + (size_t)(s0 + r) * H + kk + c);
    }
  };

  stage(0, 0);
  __syncthreads();
  const int nSteps = H / 32;
  for (int step = 0; step < nSteps; ++step) {
    const int cur = step & 1;
    v16h a = load_a_frag(&ldsQ[cur][(wave * 16 + m) * 32], half);
    v16h bf[8];
#pragma unroll
    for (int nt = 0; nt < 8; ++nt)
      bf[nt] = load_b_frag(&ldsK[cur][(nt * 16 + m) * 32], half);
#pragma unroll
    for (int nt = 0; nt < 8; ++nt) acc[nt] = wmma_f16(a, bf[nt], acc[nt]);
    if (step + 1 < nSteps) stage(cur ^ 1, (step + 1) * 32);
    __syncthreads();
  }

  const float scale = 11.3137084989847603904f;       // sqrt(128)
#pragma unroll
  for (int nt = 0; nt < 8; ++nt)
#pragma unroll
    for (int r = 0; r < 8; ++r) {
      int t = t0 + wave * 16 + r + 8 * half;
      int s = s0 + nt * 16 + m;
      float sc = acc[nt][r] * scale;
      if (s > t) sc = MASK_NEG;
      att[((size_t)b * T + t) * T + s] = sc;
    }
}

// ---------------------------------------------------------------------------
// 3) column stats over t (softmax axis): m[s]=max_t att, r[s]=1/sum_t exp.
//    One thread per (b,s), starting at its own 128-tile boundary (only written
//    tiles touched, per-wave coalesced). 4 interleaved online-softmax partials
//    cut the dependent v_exp chain 4x; merged with the standard combine.
// ---------------------------------------------------------------------------
__global__ __launch_bounds__(256) void colstats(
    const float* __restrict__ att, float* __restrict__ mb,
    float* __restrict__ rb) {
  const int idx = blockIdx.x * 256 + threadIdx.x;    // 0..B*T-1
  const int b = idx >> 11;                           // /T (T=2048)
  const int s = idx & (T - 1);
  const float* col = att + (size_t)b * T * T + s;

  float mx[4], d[4];
#pragma unroll
  for (int j = 0; j < 4; ++j) { mx[j] = MASK_NEG; d[j] = 0.f; }

  const int tStart = s & ~127;                       // wave-uniform, /4 exact
  for (int t = tStart; t < T; t += 4) {
#pragma unroll
    for (int j = 0; j < 4; ++j) {
      float a = col[(size_t)(t + j) * T];
      float mn = fmaxf(mx[j], a);
      d[j] = d[j] * __expf(mx[j] - mn) + __expf(a - mn);
      mx[j] = mn;
    }
  }
  float M = fmaxf(fmaxf(mx[0], mx[1]), fmaxf(mx[2], mx[3]));
  float D = d[0] * __expf(mx[0] - M) + d[1] * __expf(mx[1] - M) +
            d[2] * __expf(mx[2] - M) + d[3] * __expf(mx[3] - M);
  mb[idx] = M;
  rb[idx] = 1.f / D;
}

// ---------------------------------------------------------------------------
// 4) out = P @ v where P[t,s] = exp(att-m[s])*r[s]   (f16 on the fly).
//    grid=(T/128, B); (sTile,kc) flattened to step, s0 = step*32; diag tile
//    self-masks because exp(MASK_NEG - m) == 0.
// ---------------------------------------------------------------------------
__global__ __launch_bounds__(256) void attn_out(
    const float* __restrict__ att, const _Float16* __restrict__ v,
    const float* __restrict__ mb, const float* __restrict__ rb,
    float* __restrict__ out) {
  const int tTile = blockIdx.x, b = blockIdx.y;
  __shared__ _Float16 ldsP[2][128 * 32];             // [t_local][s_local]  (A)
  __shared__ _Float16 ldsVt[2][128 * 32];            // [h][s_local]       (B)

  const int t0 = tTile * 128;
  const float* attb = att + (size_t)b * T * T;
  const _Float16* vb = v + (size_t)b * T * H;
  const float* mbb = mb + b * T;
  const float* rbb = rb + b * T;
  const int tid = threadIdx.x, wave = tid >> 5, lane = tid & 31;
  const int m = lane & 15, half = lane >> 4;

  v8f acc[8];
#pragma unroll
  for (int i = 0; i < 8; ++i) acc[i] = zero8();

  auto stage = [&](int buf, int step) {
    const int s0 = step * 32;
    // P tile: float4 att + float4 stats -> 4x expf -> packed v4h LDS store.
#pragma unroll
    for (int i = 0; i < 4; ++i) {
      int chunk = tid + i * 256;                     // 0..1023 over 128x32
      int r = chunk >> 3, cc = (chunk & 7) * 4;
      int s = s0 + cc;
      float4 a  = *(const float4*)(attb + (size_t)(t0 + r) * T + s);
      float4 mv = *(const float4*)(mbb + s);
      float4 rv = *(const float4*)(rbb + s);
      __builtin_prefetch(attb + (size_t)(t0 + r) * T + s + 128, 0, 0);
      v4h p = {(_Float16)(__expf(a.x - mv.x) * rv.x),
               (_Float16)(__expf(a.y - mv.y) * rv.y),
               (_Float16)(__expf(a.z - mv.z) * rv.z),
               (_Float16)(__expf(a.w - mv.w) * rv.w)};
      *(v4h*)(&ldsP[buf][r * 32 + cc]) = p;
    }
    // V tile transposed: v8h loads coalesced over H, b16 scatter into LDS.
#pragma unroll
    for (int i = 0; i < 2; ++i) {
      int chunk = tid + i * 256;                     // 0..511 over 32x128
      int r = chunk >> 4, c = (chunk & 15) * 8;
      v8h vv = *(const v8h*)(vb + (size_t)(s0 + r) * H + c);
#pragma unroll
      for (int j = 0; j < 8; ++j) ldsVt[buf][(c + j) * 32 + r] = vv[j];
    }
  };

  const int nSteps = 4 * (tTile + 1);                // 32-wide s-chunks
  stage(0, 0);
  __syncthreads();
  for (int step = 0; step < nSteps; ++step) {
    const int cur = step & 1;
    v16h a = load_a_frag(&ldsP[cur][(wave * 16 + m) * 32], half);
    v16h bf[8];
#pragma unroll
    for (int nt = 0; nt < 8; ++nt)
      bf[nt] = load_b_frag(&ldsVt[cur][(nt * 16 + m) * 32], half);
#pragma unroll
    for (int nt = 0; nt < 8; ++nt) acc[nt] = wmma_f16(a, bf[nt], acc[nt]);
    if (step + 1 < nSteps) stage(cur ^ 1, step + 1);
    __syncthreads();
  }

#pragma unroll
  for (int nt = 0; nt < 8; ++nt)
#pragma unroll
    for (int r = 0; r < 8; ++r) {
      int t = t0 + wave * 16 + r + 8 * half;
      out[((size_t)b * T + t) * H + nt * 16 + m] = acc[nt][r];
    }
}

// ---------------------------------------------------------------------------
extern "C" void kernel_launch(void* const* d_in, const int* in_sizes, int n_in,
                              void* d_out, int out_size, void* d_ws,
                              size_t ws_size, hipStream_t stream) {
  const float* x  = (const float*)d_in[0];
  const float* Wk = (const float*)d_in[1];
  const float* Wq = (const float*)d_in[2];
  const float* Wv = (const float*)d_in[3];
  float* out = (float*)d_out;

  // workspace layout: q,k,v (f16, 4MiB each) | m,r (f32, 64KiB each) | att f32
  char* ws = (char*)d_ws;
  const size_t qkvElems = (size_t)B * T * H;
  _Float16* q = (_Float16*)ws;
  _Float16* k = q + qkvElems;
  _Float16* v = k + qkvElems;
  float* mb  = (float*)(ws + 3 * qkvElems * sizeof(_Float16));
  float* rb  = mb + (size_t)B * T;
  float* att = rb + (size_t)B * T;   // B*T*T floats = 128 MiB

  qkv_proj<<<dim3(B * T / 128, 3), 256, 0, stream>>>(x, Wq, Wk, Wv, q, k, v);
  scores<<<dim3(T / 128, T / 128, B), 256, 0, stream>>>(q, k, att);
  colstats<<<dim3(B * T / 256), 256, 0, stream>>>(att, mb, rb);
  attn_out<<<dim3(T / 128, B), 256, 0, stream>>>(att, v, mb, rb, out);
}
